// SelfAttention_56736517980684
// MI455X (gfx1250) — compile-verified, matrix operand order checked by reading, not run on hardware
//
#include <hip/hip_runtime.h>

// Problem sizes (B, M, H) = (8, 2048, 1024); N (key/attn dim) == M.
#define BB 8
#define MM 2048
#define HH 1024
#define NN 2048

typedef __attribute__((ext_vector_type(16))) __bf16 v16bf;
typedef __attribute__((ext_vector_type(8)))  __bf16 v8bf;
typedef __attribute__((ext_vector_type(8)))  float  v8f;
typedef __attribute__((ext_vector_type(4)))  float  v4f;

// Load a 16-element bf16 fragment (one lane's share of a 16x32 A / 32x16 B tile)
// from f32 memory, converting on the fly. Per-lane K pattern (ISA 7.12.2,
// 16-bit A-matrix 16x32): k = kOff + {0..7} and kOff + 16 + {0..7},
// where kOff = (lane>=16) ? 8 : 0.  p must already include row*ld + kBase + kOff.
__device__ __forceinline__ v16bf frag_from_f32(const float* __restrict__ p) {
  v4f a = *(const v4f*)(p);
  v4f b = *(const v4f*)(p + 4);
  v4f c = *(const v4f*)(p + 16);
  v4f d = *(const v4f*)(p + 20);
  v16bf r;
#pragma unroll
  for (int i = 0; i < 4; ++i) {
    r[i]      = (__bf16)a[i];
    r[4 + i]  = (__bf16)b[i];
    r[8 + i]  = (__bf16)c[i];
    r[12 + i] = (__bf16)d[i];
  }
  return r;
}

// Same fragment pattern but from bf16 memory: two 16-byte loads.
__device__ __forceinline__ v16bf frag_from_bf16(const __bf16* __restrict__ p) {
  v8bf a = *(const v8bf*)(p);
  v8bf b = *(const v8bf*)(p + 16);
  v16bf r;
#pragma unroll
  for (int i = 0; i < 8; ++i) { r[i] = a[i]; r[8 + i] = b[i]; }
  return r;
}

// ---------------------------------------------------------------------------
// Kernel 1: S[b] = Q[b] (2048x1024) * K[b]^T (1024x2048), output bf16.
// Block = 256 threads = 8 waves; wave tile 64(M) x 32(N); block tile 128x128.
// (4x2 tiles per wave: f32->bf16 staging keeps VGPR use ~130; 4x4 would spill.)
// ---------------------------------------------------------------------------
__global__ __launch_bounds__(256)
void gemm_qk_bf16(const float* __restrict__ Q, const float* __restrict__ K,
                  __bf16* __restrict__ S) {
  const int lane = threadIdx.x & 31;
  const int wave = threadIdx.x >> 5;
  const int lr   = lane & 15;
  const int kOff = (lane >> 4) << 3;           // 0 or 8
  const int b  = blockIdx.z;
  const int m0 = blockIdx.x * 128 + (wave & 1) * 64;
  const int n0 = blockIdx.y * 128 + (wave >> 1) * 32;

  const float* Qb = Q + (size_t)b * MM * HH;
  const float* Kb = K + (size_t)b * MM * HH;

  v8f acc[4][2] = {};

  const float* qrow[4];
  const float* krow[2];
#pragma unroll
  for (int i = 0; i < 4; ++i)
    qrow[i] = Qb + (size_t)(m0 + 16 * i + lr) * HH + kOff;
#pragma unroll
  for (int j = 0; j < 2; ++j)
    krow[j] = Kb + (size_t)(n0 + 16 * j + lr) * HH + kOff;

  for (int h = 0; h < HH; h += 32) {
    // Prefetch ~8 k-steps ahead (speculative; OOB prefetches are dropped).
#pragma unroll
    for (int i = 0; i < 4; ++i) __builtin_prefetch(qrow[i] + h + 256, 0, 1);
#pragma unroll
    for (int j = 0; j < 2; ++j) __builtin_prefetch(krow[j] + h + 256, 0, 1);

    v16bf afr[4], bfr[2];
#pragma unroll
    for (int i = 0; i < 4; ++i) afr[i] = frag_from_f32(qrow[i] + h);
#pragma unroll
    for (int j = 0; j < 2; ++j) bfr[j] = frag_from_f32(krow[j] + h);
#pragma unroll
    for (int i = 0; i < 4; ++i)
#pragma unroll
      for (int j = 0; j < 2; ++j)
        acc[i][j] = __builtin_amdgcn_wmma_f32_16x16x32_bf16(
            false, afr[i], false, bfr[j], (short)0, acc[i][j], false, false);
  }

  // C/D layout: row = r + 8*(lane>=16), col = lane%16
  __bf16* Sb = S + (size_t)b * MM * NN;
  const int rhi = (lane >> 4) << 3;
#pragma unroll
  for (int i = 0; i < 4; ++i)
#pragma unroll
    for (int j = 0; j < 2; ++j)
#pragma unroll
      for (int r = 0; r < 8; ++r)
        Sb[(size_t)(m0 + 16 * i + r + rhi) * NN + (n0 + 16 * j + lr)] =
            (__bf16)acc[i][j][r];
}

// ---------------------------------------------------------------------------
// Kernel 2: Vt[b][h][n] = bf16(V[b][n][h])  (transpose + downconvert via LDS)
// ---------------------------------------------------------------------------
__global__ __launch_bounds__(256)
void vt_bf16_kernel(const float* __restrict__ V, __bf16* __restrict__ Vt) {
  __shared__ __bf16 tile[64][64 + 4];
  const int b  = blockIdx.z;
  const int n0 = blockIdx.x * 64;
  const int h0 = blockIdx.y * 64;
  const float* Vb  = V  + (size_t)b * NN * HH;
  __bf16*      Vtb = Vt + (size_t)b * HH * NN;
  const int tx = threadIdx.x & 15;   // 16 groups of 4 along fast dim
  const int ty = threadIdx.x >> 4;   // 16 rows per pass
#pragma unroll
  for (int it = 0; it < 4; ++it) {
    int n = ty + it * 16;
    v4f v = *(const v4f*)(Vb + (size_t)(n0 + n) * HH + h0 + tx * 4);
#pragma unroll
    for (int k = 0; k < 4; ++k) tile[n][tx * 4 + k] = (__bf16)v[k];
  }
  __syncthreads();
#pragma unroll
  for (int it = 0; it < 4; ++it) {
    int h = ty + it * 16;
    __bf16* dst = Vtb + (size_t)(h0 + h) * NN + n0 + tx * 4;
#pragma unroll
    for (int k = 0; k < 4; ++k) dst[k] = tile[tx * 4 + k][h];
  }
}

// ---------------------------------------------------------------------------
// Kernel 3: O[b] = S[b] (bf16, 2048x2048) * V[b] (2048x1024), f32 out.
// bf16 operands need no conversion staging, so use a 64x64 wave tile
// (4x4 16x16 accumulators): 16 WMMA per 16 b128 loads per k-step.
// Block = 8 waves as 2(M) x 4(H) -> block tile 128 x 256.
// ---------------------------------------------------------------------------
__global__ __launch_bounds__(256)
void gemm_sv_bf16(const __bf16* __restrict__ S, const __bf16* __restrict__ Vt,
                  float* __restrict__ O) {
  const int lane = threadIdx.x & 31;
  const int wave = threadIdx.x >> 5;
  const int lr   = lane & 15;
  const int kOff = (lane >> 4) << 3;
  const int b  = blockIdx.z;
  const int m0 = blockIdx.x * 128 + (wave & 1) * 64;
  const int h0 = blockIdx.y * 256 + (wave >> 1) * 64;

  const __bf16* Sb  = S  + (size_t)b * MM * NN;
  const __bf16* Vtb = Vt + (size_t)b * HH * NN;

  v8f acc[4][4] = {};

  const __bf16* srow[4];
  const __bf16* vrow[4];
#pragma unroll
  for (int i = 0; i < 4; ++i)
    srow[i] = Sb + (size_t)(m0 + 16 * i + lr) * NN + kOff;
#pragma unroll
  for (int j = 0; j < 4; ++j)
    vrow[j] = Vtb + (size_t)(h0 + 16 * j + lr) * NN + kOff;

  for (int n = 0; n < NN; n += 32) {
#pragma unroll
    for (int i = 0; i < 4; ++i) __builtin_prefetch(srow[i] + n + 512, 0, 1);
#pragma unroll
    for (int j = 0; j < 4; ++j) __builtin_prefetch(vrow[j] + n + 512, 0, 1);

    v16bf afr[4], bfr[4];
#pragma unroll
    for (int i = 0; i < 4; ++i) afr[i] = frag_from_bf16(srow[i] + n);
#pragma unroll
    for (int j = 0; j < 4; ++j) bfr[j] = frag_from_bf16(vrow[j] + n);
#pragma unroll
    for (int i = 0; i < 4; ++i)
#pragma unroll
      for (int j = 0; j < 4; ++j)
        acc[i][j] = __builtin_amdgcn_wmma_f32_16x16x32_bf16(
            false, afr[i], false, bfr[j], (short)0, acc[i][j], false, false);
  }

  float* Ob = O + (size_t)b * MM * HH;
  const int rhi = (lane >> 4) << 3;
#pragma unroll
  for (int i = 0; i < 4; ++i)
#pragma unroll
    for (int j = 0; j < 4; ++j)
#pragma unroll
      for (int r = 0; r < 8; ++r)
        Ob[(size_t)(m0 + 16 * i + r + rhi) * HH + (h0 + 16 * j + lr)] =
            acc[i][j][r];
}

extern "C" void kernel_launch(void* const* d_in, const int* in_sizes, int n_in,
                              void* d_out, int out_size, void* d_ws, size_t ws_size,
                              hipStream_t stream) {
  (void)in_sizes; (void)n_in; (void)out_size; (void)ws_size;
  const float* query = (const float*)d_in[0];
  const float* key   = (const float*)d_in[1];
  // d_in[2] = span   : unused by the reference's returned value
  const float* value = (const float*)d_in[3];
  // d_in[4] = key_pe : unused by the reference's returned value
  float* out = (float*)d_out;

  // Workspace: S (bf16, 64 MB) then Vt (bf16, 32 MB)
  __bf16* Sbf = (__bf16*)d_ws;
  __bf16* Vt  = Sbf + (size_t)BB * MM * NN;

  dim3 blk(256);
  vt_bf16_kernel<<<dim3(NN / 64, HH / 64, BB), blk, 0, stream>>>(value, Vt);
  gemm_qk_bf16 <<<dim3(MM / 128, NN / 128, BB), blk, 0, stream>>>(query, key, Sbf);
  gemm_sv_bf16 <<<dim3(MM / 128, HH / 256, BB), blk, 0, stream>>>(Sbf, Vt, out);
}
